// BasicTransformerBlock_30734785970402
// MI455X (gfx1250) — matrix operand order repaired and checked by reading, not compiled, not run on hardware
//
#include <hip/hip_runtime.h>
#include <hip/hip_bf16.h>

// ---------------------------------------------------------------------------
// CDNA5 (gfx1250) wave32 WMMA implementation of the key-frame cross attention
// block.  All matmuls run through v_wmma_f32_16x16x32_f16 (f16 operands, f32
// accumulate).  Workload: B=16, S=1024, C=320, H=8, D=40.
// ---------------------------------------------------------------------------

typedef __attribute__((ext_vector_type(16))) _Float16 v16h;
typedef __attribute__((ext_vector_type(8)))  float    v8f;
typedef __attribute__((ext_vector_type(4)))  unsigned int u32x4;

#define TOKENS   16384
#define CDIM     320
#define SEQ      1024
#define BFRAMES  16
#define NHEADS   8
#define DHEAD    40
#define SCALE    0.15811388300841897f   // 40^-0.5

// Load a v16h WMMA fragment as two 16B LDS chunks (ds_load_b128 x2).
__device__ __forceinline__ v16h ld_frag16(const _Float16* p0, const _Float16* p1) {
    union { v16h v; u32x4 q[2]; } u;
    u.q[0] = *reinterpret_cast<const u32x4*>(p0);
    u.q[1] = *reinterpret_cast<const u32x4*>(p1);
    return u.v;
}

__device__ __forceinline__ v8f wmma16(v16h a, v16h b, v8f c) {
    // 8 args: (neg_a, A, neg_b, B, c_mod, C, reuse_a, reuse_b)
    return __builtin_amdgcn_wmma_f32_16x16x32_f16(false, a, false, b, (short)0, c,
                                                  false, false);
}

// Convert 8 fp32 (two float4) -> 8 f16 and store as one 16B LDS chunk.
__device__ __forceinline__ void cvt8_store(const float4 a, const float4 b, _Float16* dstp) {
    union { _Float16 h[8]; u32x4 q; } u;
    u.h[0] = (_Float16)a.x; u.h[1] = (_Float16)a.y;
    u.h[2] = (_Float16)a.z; u.h[3] = (_Float16)a.w;
    u.h[4] = (_Float16)b.x; u.h[5] = (_Float16)b.y;
    u.h[6] = (_Float16)b.z; u.h[7] = (_Float16)b.w;
    *reinterpret_cast<u32x4*>(dstp) = u.q;
}

// ---------------------------------------------------------------------------
// Kernel 1: fused Q/K/V projection.  out = hs @ W.T  (nn.Linear, no bias)
// Block tile 128(M) x 64(N), K-step 32.  8 waves, each computes 32x32.
// Grid: (16384/128) * (960/64) = 128 * 15.  nblk -> {Wq,Wk,Wv} x column group.
// ---------------------------------------------------------------------------
__global__ __launch_bounds__(256) void qkv_gemm(
        const float* __restrict__ hs,
        const float* __restrict__ Wq, const float* __restrict__ Wk,
        const float* __restrict__ Wv,
        _Float16* __restrict__ qh, _Float16* __restrict__ kh,
        _Float16* __restrict__ vh) {
    __shared__ __attribute__((aligned(16))) _Float16 a_lds[128][40]; // 32 used, pad->40
    __shared__ __attribute__((aligned(16))) _Float16 b_lds[64][40];

    const int tid  = threadIdx.x;
    const int lane = tid & 31, w = tid >> 5;
    const int m16  = lane & 15, g = lane >> 4;
    const int wm   = w & 3,  wn = w >> 2;

    const int nblk = blockIdx.x % 15, mblk = blockIdx.x / 15;
    const int which = nblk / 5;
    const int wcol  = (nblk % 5) * 64;
    const int m0    = mblk * 128;

    const float* W  = (which == 0) ? Wq : ((which == 1) ? Wk : Wv);
    _Float16*  dst  = (which == 0) ? qh : ((which == 1) ? kh : vh);

    v8f acc[2][2] = {};

    const int ar = tid >> 1, acs = (tid & 1) * 16;  // A staging: 16 halfs/thread
    const int br = tid >> 2, bcs = (tid & 3) * 8;   // B staging:  8 halfs/thread

    for (int k0 = 0; k0 < CDIM; k0 += 32) {
        __syncthreads();
        {   // stage A tile: hs rows, fp32 -> f16
            const float4* s4 = reinterpret_cast<const float4*>(
                hs + (size_t)(m0 + ar) * CDIM + k0 + acs);
            float4 f0 = s4[0], f1 = s4[1], f2 = s4[2], f3 = s4[3];
            cvt8_store(f0, f1, &a_lds[ar][acs]);
            cvt8_store(f2, f3, &a_lds[ar][acs + 8]);
        }
        {   // stage B tile: W rows are already Bt (N-major over K)
            const float4* s4 = reinterpret_cast<const float4*>(
                W + (size_t)(wcol + br) * CDIM + k0 + bcs);
            float4 f0 = s4[0], f1 = s4[1];
            cvt8_store(f0, f1, &b_lds[br][bcs]);
        }
        __syncthreads();

        v16h a0 = ld_frag16(&a_lds[wm * 32 + m16][g * 16],
                            &a_lds[wm * 32 + m16][g * 16 + 8]);
        v16h a1 = ld_frag16(&a_lds[wm * 32 + 16 + m16][g * 16],
                            &a_lds[wm * 32 + 16 + m16][g * 16 + 8]);
        v16h b0 = ld_frag16(&b_lds[wn * 32 + m16][g * 16],
                            &b_lds[wn * 32 + m16][g * 16 + 8]);
        v16h b1 = ld_frag16(&b_lds[wn * 32 + 16 + m16][g * 16],
                            &b_lds[wn * 32 + 16 + m16][g * 16 + 8]);

        acc[0][0] = wmma16(a0, b0, acc[0][0]);
        acc[0][1] = wmma16(a0, b1, acc[0][1]);
        acc[1][0] = wmma16(a1, b0, acc[1][0]);
        acc[1][1] = wmma16(a1, b1, acc[1][1]);
    }

#pragma unroll
    for (int i = 0; i < 2; ++i)
#pragma unroll
        for (int j = 0; j < 2; ++j)
#pragma unroll
            for (int r = 0; r < 8; ++r) {
                int row = m0 + wm * 32 + i * 16 + r + 8 * g;
                int col = wcol + wn * 32 + j * 16 + m16;
                dst[(size_t)row * CDIM + col] = (_Float16)acc[i][j][r];
            }
}

// ---------------------------------------------------------------------------
// Kernel 2: flash attention.  Block = (frame f, head h, 128 query rows).
// 8 waves x 16 query rows; online softmax per wave; key tiles of 32.
// Grid: 16 * 8 * 8 = 1024.
// ---------------------------------------------------------------------------
__global__ __launch_bounds__(256) void attn(
        const _Float16* __restrict__ qh, const _Float16* __restrict__ kh,
        const _Float16* __restrict__ vh, _Float16* __restrict__ oh,
        const int* __restrict__ vlp, const int* __restrict__ kp) {
    __shared__ __attribute__((aligned(16))) _Float16 q_lds[128][72]; // 64 used
    __shared__ __attribute__((aligned(16))) _Float16 k_lds[32][72];
    __shared__ __attribute__((aligned(16))) _Float16 vt_lds[48][40]; // [d][t]
    __shared__ __attribute__((aligned(16))) _Float16 p_lds[8][16][40];

    const int tid  = threadIdx.x;
    const int lane = tid & 31, w = tid >> 5;
    const int m16  = lane & 15, g = lane >> 4;

    const int qt = blockIdx.x & 7;
    const int h  = (blockIdx.x >> 3) & 7;
    const int f  = blockIdx.x >> 6;

    int vln = vlp[0]; if (vln <= 0) vln = 1;
    int kidx = kp[0]; if (kidx < 0) kidx = 0; if (kidx >= vln) kidx = vln - 1;
    int kvf = (f / vln) * vln + kidx;
    if (kvf >= BFRAMES) kvf = BFRAMES - 1;

    const size_t q_row0  = (size_t)f * SEQ + qt * 128;
    const size_t kv_row0 = (size_t)kvf * SEQ;

    // ---- stage Q tile once (scale folded in; d in [40,64) zero-padded) ----
    {
        const int qr = tid >> 1, qcs = (tid & 1) * 32;
        const _Float16* qsrc = qh + (q_row0 + qr) * CDIM + h * DHEAD;
#pragma unroll
        for (int d = 0; d < 32; ++d) {
            int dd = qcs + d;
            float v = (dd < DHEAD) ? (float)qsrc[dd] * SCALE : 0.0f;
            q_lds[qr][dd] = (_Float16)v;
        }
    }
    __syncthreads();

    // Q A-fragments, kept in registers for the whole key loop.
    v16h aq0 = ld_frag16(&q_lds[w * 16 + m16][g * 16],
                         &q_lds[w * 16 + m16][g * 16 + 8]);       // d 0..31
    v16h aq1 = ld_frag16(&q_lds[w * 16 + m16][32 + g * 16],
                         &q_lds[w * 16 + m16][32 + g * 16 + 8]);       // d 32..63

    v8f o0 = {}, o1 = {}, o2 = {};
    float mrow[8], lrow[8];
#pragma unroll
    for (int r = 0; r < 8; ++r) { mrow[r] = -3.0e38f; lrow[r] = 0.0f; }

    const int kr = tid >> 3, kcs = (tid & 7) * 8;   // K staging
    const int vtc = tid & 31, vdb = (tid >> 5) * 6; // V staging (transposed)

    for (int t0 = 0; t0 < SEQ; t0 += 32) {
        {   // stage K tile [32][64], zero-padded past DHEAD
            const _Float16* ksrc = kh + (kv_row0 + t0 + kr) * CDIM + h * DHEAD;
#pragma unroll
            for (int d = 0; d < 8; ++d) {
                int dd = kcs + d;
                k_lds[kr][dd] = (dd < DHEAD) ? ksrc[dd] : (_Float16)0.0f;
            }
        }
        {   // stage V transposed: vt_lds[d][t]
            const _Float16* vsrc = vh + (kv_row0 + t0 + vtc) * CDIM + h * DHEAD;
#pragma unroll
            for (int i = 0; i < 6; ++i) {
                int d = vdb + i;
                vt_lds[d][vtc] = (d < DHEAD) ? vsrc[d] : (_Float16)0.0f;
            }
        }
        __syncthreads();

        // ---- scores: S = Q K^T  (two 16-key fragments, two K-steps) ----
        v16h kb00 = ld_frag16(&k_lds[m16][g * 16],       &k_lds[m16][g * 16 + 8]);
        v16h kb01 = ld_frag16(&k_lds[m16][32 + g * 16],  &k_lds[m16][32 + g * 16 + 8]);
        v16h kb10 = ld_frag16(&k_lds[16 + m16][g * 16],  &k_lds[16 + m16][g * 16 + 8]);
        v16h kb11 = ld_frag16(&k_lds[16 + m16][32 + g * 16],
                              &k_lds[16 + m16][32 + g * 16 + 8]);
        v8f c0 = {}, c1 = {};
        c0 = wmma16(aq0, kb00, c0);
        c0 = wmma16(aq1, kb01, c0);
        c1 = wmma16(aq0, kb10, c1);
        c1 = wmma16(aq1, kb11, c1);

        // ---- online softmax (row m = r + 8*g lives in one 16-lane half) ----
#pragma unroll
        for (int r = 0; r < 8; ++r) {
            float s0 = c0[r], s1 = c1[r];
            float tm = fmaxf(s0, s1);
            tm = fmaxf(tm, __shfl_xor(tm, 1, 32));
            tm = fmaxf(tm, __shfl_xor(tm, 2, 32));
            tm = fmaxf(tm, __shfl_xor(tm, 4, 32));
            tm = fmaxf(tm, __shfl_xor(tm, 8, 32));
            float mnew  = fmaxf(mrow[r], tm);
            float alpha = __expf(mrow[r] - mnew);
            float p0 = __expf(s0 - mnew);
            float p1 = __expf(s1 - mnew);
            float ps = p0 + p1;
            ps += __shfl_xor(ps, 1, 32);
            ps += __shfl_xor(ps, 2, 32);
            ps += __shfl_xor(ps, 4, 32);
            ps += __shfl_xor(ps, 8, 32);
            lrow[r] = lrow[r] * alpha + ps;
            mrow[r] = mnew;
            o0[r] = o0[r] * alpha;
            o1[r] = o1[r] * alpha;
            o2[r] = o2[r] * alpha;
            int m = r + 8 * g;
            p_lds[w][m][m16]      = (_Float16)p0;
            p_lds[w][m][16 + m16] = (_Float16)p1;
        }

        // ---- O += P @ V  (P re-read as A-fragment; wave-private, no barrier)
        v16h pa  = ld_frag16(&p_lds[w][m16][g * 16], &p_lds[w][m16][g * 16 + 8]);
        v16h vb0 = ld_frag16(&vt_lds[m16][g * 16],      &vt_lds[m16][g * 16 + 8]);
        v16h vb1 = ld_frag16(&vt_lds[16 + m16][g * 16], &vt_lds[16 + m16][g * 16 + 8]);
        v16h vb2 = ld_frag16(&vt_lds[32 + m16][g * 16], &vt_lds[32 + m16][g * 16 + 8]);
        o0 = wmma16(pa, vb0, o0);
        o1 = wmma16(pa, vb1, o1);
        o2 = wmma16(pa, vb2, o2);

        __syncthreads();
    }

    // ---- normalize and store (d < 40 only) ----
#pragma unroll
    for (int r = 0; r < 8; ++r) {
        float inv = 1.0f / lrow[r];
        int m = r + 8 * g;
        _Float16* op = oh + (q_row0 + w * 16 + m) * CDIM + h * DHEAD;
        op[m16]      = (_Float16)((float)o0[r] * inv);
        op[16 + m16] = (_Float16)((float)o1[r] * inv);
        if (m16 < 8)
            op[32 + m16] = (_Float16)((float)o2[r] * inv);
    }
}

// ---------------------------------------------------------------------------
// Kernel 3: output projection.  out = O @ Wo.T + bo  (fp32 result)
// Grid: (16384/128) * (320/64) = 128 * 5.
// ---------------------------------------------------------------------------
__global__ __launch_bounds__(256) void out_gemm(
        const _Float16* __restrict__ oh, const float* __restrict__ Wo,
        const float* __restrict__ bo, float* __restrict__ out) {
    __shared__ __attribute__((aligned(16))) _Float16 a_lds[128][40];
    __shared__ __attribute__((aligned(16))) _Float16 b_lds[64][40];

    const int tid  = threadIdx.x;
    const int lane = tid & 31, w = tid >> 5;
    const int m16  = lane & 15, g = lane >> 4;
    const int wm   = w & 3,  wn = w >> 2;

    const int nblk = blockIdx.x % 5, mblk = blockIdx.x / 5;
    const int wcol = nblk * 64;
    const int m0   = mblk * 128;

    v8f acc[2][2] = {};

    const int ar = tid >> 1, acs = (tid & 1) * 16;
    const int br = tid >> 2, bcs = (tid & 3) * 8;

    for (int k0 = 0; k0 < CDIM; k0 += 32) {
        __syncthreads();
        {   // A tile is already f16 in workspace: straight 16B copies
            const u32x4* s4 = reinterpret_cast<const u32x4*>(
                oh + (size_t)(m0 + ar) * CDIM + k0 + acs);
            u32x4 d0 = s4[0], d1 = s4[1];
            *reinterpret_cast<u32x4*>(&a_lds[ar][acs])     = d0;
            *reinterpret_cast<u32x4*>(&a_lds[ar][acs + 8]) = d1;
        }
        {   // B tile from Wo (fp32 -> f16)
            const float4* s4 = reinterpret_cast<const float4*>(
                Wo + (size_t)(wcol + br) * CDIM + k0 + bcs);
            float4 f0 = s4[0], f1 = s4[1];
            cvt8_store(f0, f1, &b_lds[br][bcs]);
        }
        __syncthreads();

        v16h a0 = ld_frag16(&a_lds[wm * 32 + m16][g * 16],
                            &a_lds[wm * 32 + m16][g * 16 + 8]);
        v16h a1 = ld_frag16(&a_lds[wm * 32 + 16 + m16][g * 16],
                            &a_lds[wm * 32 + 16 + m16][g * 16 + 8]);
        v16h b0 = ld_frag16(&b_lds[wn * 32 + m16][g * 16],
                            &b_lds[wn * 32 + m16][g * 16 + 8]);
        v16h b1 = ld_frag16(&b_lds[wn * 32 + 16 + m16][g * 16],
                            &b_lds[wn * 32 + 16 + m16][g * 16 + 8]);

        acc[0][0] = wmma16(a0, b0, acc[0][0]);
        acc[0][1] = wmma16(a0, b1, acc[0][1]);
        acc[1][0] = wmma16(a1, b0, acc[1][0]);
        acc[1][1] = wmma16(a1, b1, acc[1][1]);
    }

#pragma unroll
    for (int i = 0; i < 2; ++i)
#pragma unroll
        for (int j = 0; j < 2; ++j)
#pragma unroll
            for (int r = 0; r < 8; ++r) {
                int row = m0 + wm * 32 + i * 16 + r + 8 * g;
                int col = wcol + wn * 32 + j * 16 + m16;
                out[(size_t)row * CDIM + col] = acc[i][j][r] + bo[col];
            }
}

// ---------------------------------------------------------------------------
extern "C" void kernel_launch(void* const* d_in, const int* in_sizes, int n_in,
                              void* d_out, int out_size, void* d_ws, size_t ws_size,
                              hipStream_t stream) {
    const float* hs = (const float*)d_in[0];
    const float* Wq = (const float*)d_in[1];
    const float* Wk = (const float*)d_in[2];
    const float* Wv = (const float*)d_in[3];
    const float* Wo = (const float*)d_in[4];
    const float* bo = (const float*)d_in[5];
    const int*  vlp = (const int*)d_in[6];
    const int*   kp = (const int*)d_in[7];
    float* out = (float*)d_out;

    // Workspace layout (f16): Q | K | V | attention-out, each 16384*320
    _Float16* qh = (_Float16*)d_ws;
    _Float16* kh = qh + (size_t)TOKENS * CDIM;
    _Float16* vh = kh + (size_t)TOKENS * CDIM;
    _Float16* oh = vh + (size_t)TOKENS * CDIM;

    qkv_gemm<<<dim3(128 * 15), dim3(256), 0, stream>>>(hs, Wq, Wk, Wv, qh, kh, vh);
    attn    <<<dim3(1024),     dim3(256), 0, stream>>>(qh, kh, vh, oh, vlp, kp);
    out_gemm<<<dim3(128 * 5),  dim3(256), 0, stream>>>(oh, Wo, bo, out);
}
